// PersonActivityModel_51994874085533
// MI455X (gfx1250) — matrix operand order repaired and checked by reading, not compiled, not run on hardware
//
#include <hip/hip_runtime.h>

typedef unsigned int  u32;
typedef unsigned short u16;
typedef __attribute__((ext_vector_type(16))) __bf16 v16bf;
typedef __attribute__((ext_vector_type(8)))  float  v8f;

#define BB   256
#define TT   128
#define INF  64
#define HH   512
#define NC   11
#define THREADS 512   // 16 waves

union FragBF { v16bf v; uint4 q[2]; };

__device__ __forceinline__ u16 f2bf(float f) {
  u32 u = __float_as_uint(f);
  u32 r = u + 0x7FFFu + ((u >> 16) & 1u);   // round-to-nearest-even
  return (u16)(r >> 16);
}

__device__ __forceinline__ float fsigmoid(float x) {
  return 1.0f / (1.0f + __expf(-x));
}
__device__ __forceinline__ float ftanh(float x) {
  float e = __expf(2.0f * x);
  return (e - 1.0f) / (e + 1.0f);
}

// A fragment: 16x32 bf16 from row-major LDS (rowStride in elements).
// lane row m=lane%16; chunks at K = half*8 and 16+half*8 (ISA 16-bit A layout).
__device__ __forceinline__ v16bf load_a(const u16* base, int rowStride, int k0, int lane) {
  int m = lane & 15, half = lane >> 4;
  const uint4* p = (const uint4*)(base + m * rowStride + k0 + half * 8);
  FragBF f;
  f.q[0] = p[0];        // K = k0 + half*8 .. +7
  f.q[1] = p[2];        // K = k0 + 16 + half*8 .. +7   (+32 bytes)
  return f.v;
}

// B fragment: 32x16 bf16, B[k][n] = W[n0+n][k0+k]; lane col n=lane%16 reads
// chunks at K = k0 + half*8 and k0 + 16 + half*8 (same order as load_a).
__device__ __forceinline__ v16bf load_b(const u16* W, int rowStride, int n0, int k0, int lane) {
  int n = lane & 15, half = lane >> 4;
  const uint4* p = (const uint4*)(W + (size_t)(n0 + n) * rowStride + k0 + half * 8);
  FragBF f;
  f.q[0] = p[0];
  f.q[1] = p[2];
  return f.v;
}

__device__ __forceinline__ v8f wmma_bf(v16bf a, v16bf b, v8f c) {
  return __builtin_amdgcn_wmma_f32_16x16x32_bf16(false, a, false, b, (short)0, c, false, false);
}

// ---------------- weight prep: fp32 -> bf16 row-major, bias fold ------------
__global__ void prep_kernel(const float* Wih, const float* Whh,
                            const float* W1, const float* W2, const float* Wo,
                            const float* b_ih, const float* b_hh,
                            u16* wih, u16* whh, u16* w1, u16* w2, u16* wo,
                            float* bsum) {
  int i = blockIdx.x * blockDim.x + threadIdx.x;
  int stride = gridDim.x * blockDim.x;
  for (int k = i; k < 4 * HH * INF; k += stride) wih[k] = f2bf(Wih[k]);
  for (int k = i; k < 4 * HH * HH;  k += stride) whh[k] = f2bf(Whh[k]);
  for (int k = i; k < HH * HH; k += stride) {
    w1[k] = f2bf(W1[k]);
    w2[k] = f2bf(W2[k]);
    wo[k] = f2bf(Wo[k]);
  }
  for (int k = i; k < 4 * HH; k += stride) bsum[k] = b_ih[k] + b_hh[k];
}

// ---------------- persistent ODE-LSTM kernel: 16 batch rows / block ---------
// LDS layout (dynamic):
//   H32  [16][512] f32   hidden state (= RK4 y)
//   C32  [16][512] f32   cell state
//   ACC  [16][512] f32   RK4 accumulator (aliased as OB for out-projection)
//   HB   [16][512] bf16  bf16 mirror of current matmul input
//   T1B  [16][512] bf16  f_node intermediate tanh(y@W1^T+b1)
//   XB   [16][64]  bf16  x slice
//   DT   [16]      f32   dt per row
struct Smem {
  float* H32; float* C32; float* ACC; float* OB;
  u16* HB; u16* T1B; u16* XB; float* DT;
};

__device__ __forceinline__ void fnode_stage(int mode, const Smem& s,
                                            const u16* w1, const float* b1,
                                            const u16* w2, const float* b2,
                                            int w, int lane) {
  int n_lane = lane & 15, halfm = lane >> 4;
  int n0 = w * 32;                  // wave owns cols [n0, n0+32)
  int j0 = n0 + n_lane, j1 = n0 + 16 + n_lane;

  // mm1: T1 = tanh(HB @ W1^T + b1) — both tiles share one A fragment per K step
  {
    v8f acc0 = {}, acc1 = {};
#pragma unroll 4
    for (int k0 = 0; k0 < HH; k0 += 32) {
      v16bf a = load_a(s.HB, HH, k0, lane);
      acc0 = wmma_bf(a, load_b(w1, HH, n0,      k0, lane), acc0);
      acc1 = wmma_bf(a, load_b(w1, HH, n0 + 16, k0, lane), acc1);
    }
    float bb0 = b1[j0], bb1 = b1[j1];
#pragma unroll
    for (int r = 0; r < 8; ++r) {
      int m = halfm * 8 + r;
      s.T1B[m * HH + j0] = f2bf(ftanh(acc0[r] + bb0));
      s.T1B[m * HH + j1] = f2bf(ftanh(acc1[r] + bb1));
    }
  }
  __syncthreads();

  // mm2: k = T1B @ W2^T + b2, fused RK4 epilogue
  {
    v8f acc0 = {}, acc1 = {};
#pragma unroll 4
    for (int k0 = 0; k0 < HH; k0 += 32) {
      v16bf a = load_a(s.T1B, HH, k0, lane);
      acc0 = wmma_bf(a, load_b(w2, HH, n0,      k0, lane), acc0);
      acc1 = wmma_bf(a, load_b(w2, HH, n0 + 16, k0, lane), acc1);
    }
    float bb0 = b2[j0], bb1 = b2[j1];
#pragma unroll
    for (int r = 0; r < 8; ++r) {
      int m = halfm * 8 + r;
      float dt = s.DT[m];
#pragma unroll
      for (int tt = 0; tt < 2; ++tt) {
        int idx = m * HH + (tt ? j1 : j0);
        float kv = (tt ? acc1[r] + bb1 : acc0[r] + bb0);
        float y = s.H32[idx];
        float nb;
        if (mode == 0)      { s.ACC[idx] = kv;          nb = y + 0.5f * dt * kv; }
        else if (mode == 1) { s.ACC[idx] += 2.0f * kv;  nb = y + 0.5f * dt * kv; }
        else if (mode == 2) { s.ACC[idx] += 2.0f * kv;  nb = y + dt * kv; }
        else { float sm = s.ACC[idx] + kv;
               float yn = y + dt * sm * (1.0f / 6.0f);
               s.H32[idx] = yn; nb = yn; }
        s.HB[idx] = f2bf(nb);   // next f_node input (HB not read during mm2)
      }
    }
  }
  __syncthreads();
}

__global__ void __launch_bounds__(THREADS)
odelstm_kernel(const float* __restrict__ x, const float* __restrict__ tin,
               const u16* __restrict__ wih, const u16* __restrict__ whh,
               const float* __restrict__ bsum,
               const u16* __restrict__ w1, const float* __restrict__ b1,
               const u16* __restrict__ w2, const float* __restrict__ b2,
               const u16* __restrict__ wo, const float* __restrict__ bo,
               const float* __restrict__ Wc, const float* __restrict__ bc,
               float* __restrict__ out) {
  extern __shared__ char smraw[];
  Smem s;
  s.H32 = (float*)smraw;
  s.C32 = s.H32 + 16 * HH;
  s.ACC = s.C32 + 16 * HH;
  s.OB  = s.ACC;                      // out-projection buffer aliases ACC
  s.HB  = (u16*)(s.ACC + 16 * HH);
  s.T1B = s.HB + 16 * HH;
  s.XB  = s.T1B + 16 * HH;
  s.DT  = (float*)(s.XB + 16 * INF);

  int tid = threadIdx.x;
  int lane = tid & 31;
  int w = tid >> 5;
  int b0 = blockIdx.x * 16;
  int n_lane = lane & 15, halfm = lane >> 4;

  for (int i = tid; i < 16 * HH; i += THREADS) {
    s.H32[i] = 0.0f; s.C32[i] = 0.0f; s.HB[i] = 0;
  }
  __syncthreads();

  for (int step = 0; step < TT; ++step) {
    // stage x slice (bf16) and dt
    for (int i = tid; i < 16 * INF; i += THREADS) {
      int m = i >> 6, k = i & 63;
      s.XB[i] = f2bf(x[((size_t)(b0 + m) * TT + step) * INF + k]);
    }
    if (tid < 16) s.DT[tid] = tin[(size_t)(b0 + tid) * TT + step] * (1.0f / 3.0f);
    __syncthreads();

    // ---- LSTM cell: all 4 gates of a hidden tile in one wave ----
    for (int tt = 0; tt < 2; ++tt) {
      int j0 = (w * 2 + tt) * 16;
      v8f ai = {}, af = {}, ag = {}, ao = {};
#pragma unroll
      for (int k0 = 0; k0 < INF; k0 += 32) {
        v16bf a = load_a(s.XB, INF, k0, lane);
        ai = wmma_bf(a, load_b(wih, INF, j0,          k0, lane), ai);
        af = wmma_bf(a, load_b(wih, INF, HH + j0,     k0, lane), af);
        ag = wmma_bf(a, load_b(wih, INF, 2 * HH + j0, k0, lane), ag);
        ao = wmma_bf(a, load_b(wih, INF, 3 * HH + j0, k0, lane), ao);
      }
#pragma unroll 2
      for (int k0 = 0; k0 < HH; k0 += 32) {
        v16bf a = load_a(s.HB, HH, k0, lane);
        ai = wmma_bf(a, load_b(whh, HH, j0,          k0, lane), ai);
        af = wmma_bf(a, load_b(whh, HH, HH + j0,     k0, lane), af);
        ag = wmma_bf(a, load_b(whh, HH, 2 * HH + j0, k0, lane), ag);
        ao = wmma_bf(a, load_b(whh, HH, 3 * HH + j0, k0, lane), ao);
      }
      int j = j0 + n_lane;
      float bi = bsum[j], bf_ = bsum[HH + j], bg = bsum[2 * HH + j], bo_ = bsum[3 * HH + j];
#pragma unroll
      for (int r = 0; r < 8; ++r) {
        int m = halfm * 8 + r;
        int idx = m * HH + j;
        float iv = fsigmoid(ai[r] + bi);
        float fv = fsigmoid(af[r] + bf_);
        float gv = ftanh(ag[r] + bg);
        float ov = fsigmoid(ao[r] + bo_);
        float cN = fv * s.C32[idx] + iv * gv;
        s.C32[idx] = cN;
        s.H32[idx] = ov * ftanh(cN);   // HB refreshed after barrier (old h still read)
      }
    }
    __syncthreads();
    for (int i = tid; i < 16 * HH; i += THREADS) s.HB[i] = f2bf(s.H32[i]);
    __syncthreads();

    // ---- 3 RK4 unfolds, 4 f_node stages each ----
    for (int u = 0; u < 3; ++u) {
      fnode_stage(0, s, w1, b1, w2, b2, w, lane);
      fnode_stage(1, s, w1, b1, w2, b2, w, lane);
      fnode_stage(2, s, w1, b1, w2, b2, w, lane);
      fnode_stage(3, s, w1, b1, w2, b2, w, lane);
    }

    // ---- per-step output projection: OB = h @ Wo^T + bo ----
    {
      int n0 = w * 32;
      int j0 = n0 + n_lane, j1 = n0 + 16 + n_lane;
      v8f acc0 = {}, acc1 = {};
#pragma unroll 4
      for (int k0 = 0; k0 < HH; k0 += 32) {
        v16bf a = load_a(s.HB, HH, k0, lane);
        acc0 = wmma_bf(a, load_b(wo, HH, n0,      k0, lane), acc0);
        acc1 = wmma_bf(a, load_b(wo, HH, n0 + 16, k0, lane), acc1);
      }
      float bb0 = bo[j0], bb1 = bo[j1];
#pragma unroll
      for (int r = 0; r < 8; ++r) {
        int m = halfm * 8 + r;
        s.OB[m * HH + j0] = acc0[r] + bb0;
        s.OB[m * HH + j1] = acc1[r] + bb1;
      }
    }
    __syncthreads();

    // ---- fused class head (fp32, float4 dots): [16,512]@[512,11] ----
    if (tid < 16 * NC) {
      int m = tid / NC, cc = tid % NC;
      const float4* wr = (const float4*)(Wc + (size_t)cc * HH);
      const float4* ob = (const float4*)(s.OB + m * HH);
      float acc = bc[cc];
#pragma unroll 8
      for (int q = 0; q < HH / 4; ++q) {
        float4 a = ob[q], b = wr[q];
        acc += a.x * b.x + a.y * b.y + a.z * b.z + a.w * b.w;
      }
      out[((size_t)(b0 + m) * TT + step) * NC + cc] = acc;
    }
    __syncthreads();   // OB (==ACC) reused next step
  }
}

// ---------------------------------------------------------------------------
extern "C" void kernel_launch(void* const* d_in, const int* in_sizes, int n_in,
                              void* d_out, int out_size, void* d_ws, size_t ws_size,
                              hipStream_t stream) {
  const float* x    = (const float*)d_in[0];
  const float* tin  = (const float*)d_in[1];
  const float* Wih  = (const float*)d_in[2];
  const float* Whh  = (const float*)d_in[3];
  const float* b_ih = (const float*)d_in[4];
  const float* b_hh = (const float*)d_in[5];
  const float* W1   = (const float*)d_in[6];
  const float* b1   = (const float*)d_in[7];
  const float* W2   = (const float*)d_in[8];
  const float* b2   = (const float*)d_in[9];
  const float* Wo   = (const float*)d_in[10];
  const float* bo   = (const float*)d_in[11];
  const float* Wc   = (const float*)d_in[12];
  const float* bc   = (const float*)d_in[13];
  float* out = (float*)d_out;

  // workspace layout (bytes)
  char* ws = (char*)d_ws;
  u16* wih  = (u16*)(ws + 0);                      // 2048*64  bf16
  u16* whh  = (u16*)(ws + 262144);                 // 2048*512 bf16
  u16* w1b  = (u16*)(ws + 2359296);                // 512*512  bf16
  u16* w2b  = (u16*)(ws + 2883584);
  u16* wob  = (u16*)(ws + 3407872);
  float* bsum = (float*)(ws + 3932160);            // 2048 f32

  prep_kernel<<<1024, 256, 0, stream>>>(Wih, Whh, W1, W2, Wo, b_ih, b_hh,
                                        wih, whh, w1b, w2b, wob, bsum);

  // dynamic LDS: 3*32KB f32 + 2*16KB bf16 + 2KB x + 64B dt
  size_t smem = (size_t)16 * HH * 4 * 3 + (size_t)16 * HH * 2 * 2 + 16 * INF * 2 + 64;
  (void)hipFuncSetAttribute((const void*)odelstm_kernel,
                            hipFuncAttributeMaxDynamicSharedMemorySize, (int)smem);
  odelstm_kernel<<<BB / 16, THREADS, smem, stream>>>(
      x, tin, wih, whh, bsum, w1b, b1, w2b, b2, wob, bo, Wc, bc, out);
}
